// MixtralAttention_heavy_hitter_7584912245149
// MI455X (gfx1250) — matrix operand order repaired and checked by reading, not compiled, not run on hardware
//
#include <hip/hip_runtime.h>
#include <hip/hip_bf16.h>

// ---------------------------------------------------------------------------
// MI455X (gfx1250) heavy-hitter attention, bf16 WMMA pipeline + TDM staging.
// wave32; WMMA 16x16x32 bf16 -> f32 accumulate for every matmul.
// ---------------------------------------------------------------------------

typedef __bf16 bf16_t;
typedef __bf16 v16bf __attribute__((ext_vector_type(16)));
typedef __bf16 v8bf  __attribute__((ext_vector_type(8)));
typedef float  v8f   __attribute__((ext_vector_type(8)));
typedef unsigned int u32x4 __attribute__((ext_vector_type(4)));
typedef int          i32x4 __attribute__((ext_vector_type(4)));
typedef int          i32x8 __attribute__((ext_vector_type(8)));

#define SQ    2048
#define HIDN  4096
#define NHD   32
#define NKVH  8
#define HDIM  128
#define NREP  4
#define HCNT  256
#define NEGF  (-3.402823466e38f)

// D = A(16x32) * B(32x16) + C, bf16 inputs, f32 accum
static __device__ inline v8f wmma_bf16(v16bf a, v16bf b, v8f c) {
  return __builtin_amdgcn_wmma_f32_16x16x32_bf16(false, a, false, b, (short)0, c,
                                                 false, false);
}

// A fragment: source row-major [M x K] (lda in elements).
// lane m = l%16, half = l/16; elem i -> K = k0 + half*16 + i. One 32-byte chunk.
static __device__ inline v16bf load_a_frag(const bf16_t* A, int lda, int m0,
                                           int k0, int lane) {
  int m = m0 + (lane & 15);
  int half = lane >> 4;
  const bf16_t* p = A + (size_t)m * lda + k0 + half * 16;
  return *(const v16bf*)(p);
}

// B fragment for NT gemm: source row-major [N x K] (W layout), B[k][n]=src[n][k].
// lane n = l%16, half = l/16; elem i -> K = k0 + half*16 + i. One 32-byte chunk.
static __device__ inline v16bf load_b_frag(const bf16_t* B, int ldb, int n0,
                                           int k0, int lane) {
  int n = n0 + (lane & 15);
  int half = lane >> 4;
  return *(const v16bf*)(B + (size_t)n * ldb + k0 + half * 16);
}

// ---------------------------------------------------------------------------
// elementwise f32 -> bf16
__global__ void cvt_bf16_kernel(const float* __restrict__ src,
                                bf16_t* __restrict__ dst, int count) {
  int i = blockIdx.x * blockDim.x + threadIdx.x;
  if (i < count) dst[i] = (bf16_t)src[i];
}

// ---------------------------------------------------------------------------
// C[M x N](f32) = A[M x K](bf16, row-major) @ B[N x K](bf16, row-major)^T
// block = 4 waves; each wave does a 16x64 tile; block covers 64x64.
__global__ __launch_bounds__(128) void gemm_nt_bf16_kernel(
    const bf16_t* __restrict__ A, const bf16_t* __restrict__ B,
    float* __restrict__ C, int M, int N, int K, int lda, int ldb, int ldc) {
  int lane = threadIdx.x & 31;
  int wave = threadIdx.x >> 5;
  int m0 = blockIdx.y * 64 + wave * 16;
  int n0 = blockIdx.x * 64;
  if (m0 >= M) return;

  v8f acc[4];
#pragma unroll
  for (int j = 0; j < 4; ++j)
#pragma unroll
    for (int r = 0; r < 8; ++r) acc[j][r] = 0.0f;

  for (int k0 = 0; k0 < K; k0 += 32) {
    // pull next tiles toward L2 (weights live in the 192MB L2 on MI455X)
    if (k0 + 256 < K) {
      __builtin_prefetch(
          (const void*)(A + (size_t)(m0 + (lane & 15)) * lda + k0 + 256), 0, 1);
      __builtin_prefetch(
          (const void*)(B + (size_t)(n0 + (lane & 15)) * ldb + k0 + 256), 0, 1);
    }
    v16bf a = load_a_frag(A, lda, m0, k0, lane);
#pragma unroll
    for (int j = 0; j < 4; ++j) {
      v16bf b = load_b_frag(B, ldb, n0 + j * 16, k0, lane);
      acc[j] = wmma_bf16(a, b, acc[j]);
    }
  }
  int n = lane & 15, hh = lane >> 4;
#pragma unroll
  for (int j = 0; j < 4; ++j)
#pragma unroll
    for (int r = 0; r < 8; ++r)
      C[(size_t)(m0 + hh * 8 + r) * ldc + n0 + j * 16 + n] = acc[j][r];
}

// ---------------------------------------------------------------------------
// RoPE: src f32 [S x (nheads*128)] -> dst bf16 same layout.
// one thread per (pos, head, d<64) pair.
__global__ void rope_kernel(const float* __restrict__ src,
                            bf16_t* __restrict__ dst, int nheads) {
  int i = blockIdx.x * blockDim.x + threadIdx.x;
  int total = SQ * nheads * 64;
  if (i >= total) return;
  int d = i & 63;
  int t = i >> 6;
  int h = t % nheads;
  int pos = t / nheads;
  int ld = nheads * HDIM;
  float inv = __powf(1000000.0f, -((float)(2 * d) / 128.0f));
  float fr = (float)pos * inv;
  float c = __cosf(fr), s = __sinf(fr);
  size_t base = (size_t)pos * ld + h * HDIM + d;
  float x0 = src[base];
  float x1 = src[base + 64];
  dst[base]      = (bf16_t)(x0 * c - x1 * s);
  dst[base + 64] = (bf16_t)(x1 * c + x0 * s);
}

// ---------------------------------------------------------------------------
// Build 4-bit pseudo-quantized guide features for q and k, padded to K=32.
// gq/gk layout: [NH][S][32] bf16 (cols 16..31 zero).
__global__ void gquant_kernel(const bf16_t* __restrict__ qr,
                              const bf16_t* __restrict__ kr,
                              const int* __restrict__ sorted_channel,
                              bf16_t* __restrict__ gq,
                              bf16_t* __restrict__ gk) {
  int i = blockIdx.x * blockDim.x + threadIdx.x;
  if (i >= NHD * SQ) return;
  int pos = i & (SQ - 1);
  int h = i / SQ;
  int hkv = h >> 2;  // NREP = 4

  float v[16];
  for (int pass = 0; pass < 2; ++pass) {
    for (int j = 0; j < 16; ++j) {
      int idx = sorted_channel[h * HDIM + j];
      if (pass == 0)
        v[j] = (float)qr[(size_t)pos * HIDN + h * HDIM + idx];
      else
        v[j] = (float)kr[(size_t)pos * (NKVH * HDIM) + hkv * HDIM + idx];
    }
    float mn = v[0], mx = v[0];
    for (int j = 1; j < 16; ++j) {
      mn = fminf(mn, v[j]);
      mx = fmaxf(mx, v[j]);
    }
    float rng = mx - mn;
    if (rng == 0.0f) rng = 1.0f;
    float scale = 15.0f / rng;
    bf16_t* dst = (pass == 0 ? gq : gk) + ((size_t)h * SQ + pos) * 32;
    for (int j = 0; j < 16; ++j) {
      float q = fminf(fmaxf(rintf((v[j] - mn) * scale), 0.0f), 15.0f);
      dst[j] = (bf16_t)(q / scale + mn);
    }
    for (int j = 16; j < 32; ++j) dst[j] = (bf16_t)0.0f;
  }
}

// ---------------------------------------------------------------------------
// Flash attention with heavy-hitter gating.
// grid = (S/16, NH), block = 1 wave (32 threads).
// V tiles are staged into LDS by the Tensor Data Mover (TDM), overlapped with
// the score WMMAs and synchronized with TENSORcnt.
__global__ __launch_bounds__(32) void attn_kernel(
    const bf16_t* __restrict__ qr, const bf16_t* __restrict__ kr,
    const bf16_t* __restrict__ vb, const bf16_t* __restrict__ gq,
    const bf16_t* __restrict__ gk, bf16_t* __restrict__ ao) {
  __shared__ __align__(32) float  sg[16 * SQ];      // masked gattn rows (128 KB)
  __shared__ __align__(32) bf16_t sp[16 * 32];      // p staging tile
  __shared__ __align__(128) bf16_t sv[32 * HDIM];   // TDM-staged V tile (8 KB)
  __shared__ float skth[16];

  int lane = threadIdx.x;
  int n = lane & 15, hh = lane >> 4;
  int h = blockIdx.y;
  int q0 = blockIdx.x * 16;
  int hkv = h >> 2;

  // ---- Phase 1: guide attention rows + causal mask into LDS --------------
  const bf16_t* gqh = gq + (size_t)h * SQ * 32;
  const bf16_t* gkh = gk + (size_t)h * SQ * 32;
  v16bf ga = load_a_frag(gqh, 32, q0, 0, lane);
  for (int kt = 0; kt < SQ / 16; ++kt) {
    v16bf gb = load_b_frag(gkh, 32, kt * 16, 0, lane);
    v8f g;
#pragma unroll
    for (int r = 0; r < 8; ++r) g[r] = 0.0f;
    g = wmma_bf16(ga, gb, g);
#pragma unroll
    for (int r = 0; r < 8; ++r) {
      int row = hh * 8 + r;
      int col = kt * 16 + n;
      float val = g[r] * 0.25f;                    // 1/sqrt(16)
      if (col > q0 + row) val += NEGF;             // causal mask
      sg[row * SQ + col] = val;
    }
  }
  __syncthreads();

  // ---- per-row 256th-largest via bitwise binary search (ordered uints) ---
  for (int row = 0; row < 16; ++row) {
    unsigned ans = 0u;
    for (int bit = 31; bit >= 0; --bit) {
      unsigned cand = ans | (1u << bit);
      int cnt = 0;
      for (int c = lane; c < SQ; c += 32) {
        unsigned u = __float_as_uint(sg[row * SQ + c]);
        u = (u & 0x80000000u) ? ~u : (u | 0x80000000u);
        cnt += (u >= cand) ? 1 : 0;
      }
#pragma unroll
      for (int off = 16; off > 0; off >>= 1) cnt += __shfl_xor(cnt, off, 32);
      if (cnt >= HCNT) ans = cand;
    }
    if (lane == 0) {
      unsigned b = (ans & 0x80000000u) ? (ans & 0x7FFFFFFFu) : ~ans;
      skth[row] = __uint_as_float(b);
    }
  }
  __syncthreads();

  // ---- Phase 2: online-softmax flash loop --------------------------------
  const bf16_t* qh = qr + (size_t)h * HDIM;    // row stride HIDN
  const bf16_t* kh = kr + (size_t)hkv * HDIM;  // row stride NKVH*HDIM
  const bf16_t* vh = vb + (size_t)hkv * HDIM;

  v8f o[8];
#pragma unroll
  for (int j = 0; j < 8; ++j)
#pragma unroll
    for (int r = 0; r < 8; ++r) o[j][r] = 0.0f;
  float rmax[8], rsum[8], kthv[8];
#pragma unroll
  for (int r = 0; r < 8; ++r) {
    rmax[r] = NEGF;
    rsum[r] = 0.0f;
    kthv[r] = skth[hh * 8 + r];
  }

  v16bf qa[4];
#pragma unroll
  for (int kk = 0; kk < 4; ++kk)
    qa[kk] = load_a_frag(qh, HIDN, q0, kk * 32, lane);

  const float scale = 0.08838834764831845f;  // 1/sqrt(128)
  const unsigned sv_lds = (unsigned)(size_t)(&sv[0]);

  for (int kt2 = 0; kt2 < SQ / 32; ++kt2) {
    // ---- TDM: DMA the 32x128 bf16 V tile into LDS, overlapped with scores.
    // Guard: previous iteration's LDS reads must retire before overwrite.
    asm volatile("s_wait_dscnt 0x0" ::: "memory");
    {
      unsigned long long gaddr =
          (unsigned long long)(size_t)(vh + (size_t)(kt2 * 32) * (NKVH * HDIM));
      u32x4 g0;
      g0.x = 1u;  // count=1, user descriptor, no gather
      g0.y = sv_lds;
      g0.z = (unsigned)(gaddr & 0xFFFFFFFFu);
      g0.w = (unsigned)((gaddr >> 32) & 0x01FFFFFFu) | (2u << 30);  // type=2
      i32x8 g1;
      g1[0] = 0x00010000;                 // data_size = 2 bytes
      g1[1] = (int)(1024u << 16);         // tensor_dim0 = 1024 (row length)
      g1[2] = (int)(2048u << 16);         // tensor_dim1 = 2048 (rows)
      g1[3] = (int)((unsigned)HDIM << 16);// tile_dim0 = 128
      g1[4] = 32;                         // tile_dim1 = 32 rows
      g1[5] = 1024;                       // tensor_dim0_stride = 1024
      g1[6] = 0;
      g1[7] = 0;
      i32x4 z4 = {0, 0, 0, 0};
      i32x8 z8 = {0, 0, 0, 0, 0, 0, 0, 0};
      __builtin_amdgcn_tensor_load_to_lds(g0, g1, z4, z4, z8, 0);
    }

    // ---- scores for this 32-wide kv chunk (runs while TDM is in flight) --
    float p0[8], p1[8];
#pragma unroll
    for (int t = 0; t < 2; ++t) {
      int kb = kt2 * 32 + t * 16;
      v8f s;
#pragma unroll
      for (int r = 0; r < 8; ++r) s[r] = 0.0f;
#pragma unroll
      for (int kk = 0; kk < 4; ++kk) {
        v16bf b = load_b_frag(kh, NKVH * HDIM, kb, kk * 32, lane);
        s = wmma_bf16(qa[kk], b, s);
      }
#pragma unroll
      for (int r = 0; r < 8; ++r) {
        int row = hh * 8 + r;
        int col = kb + n;
        float val = s[r] * scale;
        if (col > q0 + row) val += NEGF;           // causal
        float gv = sg[row * SQ + col];
        if (!(gv >= kthv[r])) val = -INFINITY;     // heavy-hitter gate
        if (t == 0) p0[r] = val; else p1[r] = val;
      }
    }
    // running softmax state
#pragma unroll
    for (int r = 0; r < 8; ++r) {
      float cm = fmaxf(p0[r], p1[r]);
#pragma unroll
      for (int off = 8; off > 0; off >>= 1)
        cm = fmaxf(cm, __shfl_xor(cm, off, 32));   // within 16-lane half
      float nm = fmaxf(rmax[r], cm);
      float corr = __expf(rmax[r] - nm);
      rsum[r] *= corr;
#pragma unroll
      for (int j = 0; j < 8; ++j) o[j][r] *= corr;
      rmax[r] = nm;
      p0[r] = __expf(p0[r] - nm);
      p1[r] = __expf(p1[r] - nm);
      float cs = p0[r] + p1[r];
#pragma unroll
      for (int off = 8; off > 0; off >>= 1) cs += __shfl_xor(cs, off, 32);
      rsum[r] += cs;
    }
    // stage p (16 rows x 32 kv) to LDS in row-major for A-frag reload
#pragma unroll
    for (int r = 0; r < 8; ++r) {
      int row = hh * 8 + r;
      sp[row * 32 + n]      = (bf16_t)p0[r];
      sp[row * 32 + 16 + n] = (bf16_t)p1[r];
    }
    // V tile must be resident before LDS reads
    __builtin_amdgcn_s_wait_tensorcnt(0);
    __syncthreads();
    v16bf pa = load_a_frag(sp, 32, 0, 0, lane);
    // p @ v : v tile now in LDS, row-major [kv_local][dim]
#pragma unroll
    for (int j = 0; j < 8; ++j) {
      v16bf vf;
#pragma unroll
      for (int i = 0; i < 16; ++i) {
        vf[i] = sv[(hh * 16 + i) * HDIM + j * 16 + n];
      }
      o[j] = wmma_bf16(pa, vf, o[j]);
    }
    __syncthreads();
  }

  // ---- finalize: divide by rowsum, write bf16 [pos][h*128+d] -------------
#pragma unroll
  for (int r = 0; r < 8; ++r) {
    float inv = (rsum[r] > 0.0f) ? (1.0f / rsum[r]) : 0.0f;
#pragma unroll
    for (int j = 0; j < 8; ++j)
      ao[(size_t)(q0 + hh * 8 + r) * HIDN + h * HDIM + j * 16 + n] =
          (bf16_t)(o[j][r] * inv);
  }
}

// ---------------------------------------------------------------------------
extern "C" void kernel_launch(void* const* d_in, const int* in_sizes, int n_in,
                              void* d_out, int out_size, void* d_ws,
                              size_t ws_size, hipStream_t stream) {
  (void)in_sizes; (void)n_in; (void)out_size; (void)ws_size;
  const float* hs = (const float*)d_in[0];
  // d_in[1] attention_mask (causal, recomputed inline), d_in[2] position_ids
  const int* sorted_channel = (const int*)d_in[3];
  const float* Wq = (const float*)d_in[4];
  const float* Wk = (const float*)d_in[5];
  const float* Wv = (const float*)d_in[6];
  const float* Wo = (const float*)d_in[7];
  float* out = (float*)d_out;

  char* ws = (char*)d_ws;
  size_t off = 0;
  auto carve = [&](size_t bytes) { char* p = ws + off; off += (bytes + 255) & ~(size_t)255; return p; };
  bf16_t* hsb = (bf16_t*)carve((size_t)SQ * HIDN * 2);
  bf16_t* wqb = (bf16_t*)carve((size_t)HIDN * HIDN * 2);   // reused for Wo later
  bf16_t* wkb = (bf16_t*)carve((size_t)NKVH * HDIM * HIDN * 2);
  bf16_t* wvb = (bf16_t*)carve((size_t)NKVH * HDIM * HIDN * 2);
  float*  qf  = (float*)carve((size_t)SQ * HIDN * 4);
  float*  kf  = (float*)carve((size_t)SQ * NKVH * HDIM * 4);
  float*  vf  = (float*)carve((size_t)SQ * NKVH * HDIM * 4);
  bf16_t* qr  = (bf16_t*)carve((size_t)SQ * HIDN * 2);
  bf16_t* kr  = (bf16_t*)carve((size_t)SQ * NKVH * HDIM * 2);
  bf16_t* vbb = (bf16_t*)carve((size_t)SQ * NKVH * HDIM * 2);
  bf16_t* gq  = (bf16_t*)carve((size_t)NHD * SQ * 32 * 2);
  bf16_t* gk  = (bf16_t*)carve((size_t)NHD * SQ * 32 * 2);
  bf16_t* ao  = (bf16_t*)carve((size_t)SQ * HIDN * 2);

  const int CT = 256;
  int nhs = SQ * HIDN;
  int nwq = HIDN * HIDN;
  int nwk = NKVH * HDIM * HIDN;
  cvt_bf16_kernel<<<(nhs + CT - 1) / CT, CT, 0, stream>>>(hs, hsb, nhs);
  cvt_bf16_kernel<<<(nwq + CT - 1) / CT, CT, 0, stream>>>(Wq, wqb, nwq);
  cvt_bf16_kernel<<<(nwk + CT - 1) / CT, CT, 0, stream>>>(Wk, wkb, nwk);
  cvt_bf16_kernel<<<(nwk + CT - 1) / CT, CT, 0, stream>>>(Wv, wvb, nwk);

  // Q/K/V projections
  gemm_nt_bf16_kernel<<<dim3(HIDN / 64, SQ / 64), 128, 0, stream>>>(
      hsb, wqb, qf, SQ, HIDN, HIDN, HIDN, HIDN, HIDN);
  gemm_nt_bf16_kernel<<<dim3((NKVH * HDIM) / 64, SQ / 64), 128, 0, stream>>>(
      hsb, wkb, kf, SQ, NKVH * HDIM, HIDN, HIDN, HIDN, NKVH * HDIM);
  gemm_nt_bf16_kernel<<<dim3((NKVH * HDIM) / 64, SQ / 64), 128, 0, stream>>>(
      hsb, wvb, vf, SQ, NKVH * HDIM, HIDN, HIDN, HIDN, NKVH * HDIM);

  // RoPE q/k, convert v
  int nq = SQ * NHD * 64, nk = SQ * NKVH * 64;
  rope_kernel<<<(nq + CT - 1) / CT, CT, 0, stream>>>(qf, qr, NHD);
  rope_kernel<<<(nk + CT - 1) / CT, CT, 0, stream>>>(kf, kr, NKVH);
  int nv = SQ * NKVH * HDIM;
  cvt_bf16_kernel<<<(nv + CT - 1) / CT, CT, 0, stream>>>(vf, vbb, nv);

  // guide features
  gquant_kernel<<<(NHD * SQ + CT - 1) / CT, CT, 0, stream>>>(qr, kr,
                                                             sorted_channel,
                                                             gq, gk);

  // convert Wo into the (now free) Wq slot
  cvt_bf16_kernel<<<(nwq + CT - 1) / CT, CT, 0, stream>>>(Wo, wqb, nwq);

  // attention
  attn_kernel<<<dim3(SQ / 16, NHD), 32, 0, stream>>>(qr, kr, vbb, gq, gk, ao);

  // output projection -> f32 result
  gemm_nt_bf16_kernel<<<dim3(HIDN / 64, SQ / 64), 128, 0, stream>>>(
      ao, wqb, out, SQ, HIDN, HIDN, HIDN, HIDN, HIDN);
}
